// BasicBlock_1w1a_49460843381707
// MI455X (gfx1250) — 8-bit matrix operand order repaired and checked by reading, compiled, not run on hardware
//
#include <hip/hip_runtime.h>
#include <hip/hip_bf16.h>

typedef int v8i __attribute__((ext_vector_type(8)));
typedef int gv4i __attribute__((vector_size(16)));

#define NIMG   64
#define CCH    256
#define HDIM   32
#define WDIM   32
#define NPIX   (NIMG * HDIM * WDIM)
#define KTOT   (CCH * 9)
#define KSTEPS 36
#define NTILES (CCH / 16)
#define MTILES (NPIX / 16)

#define PDIM   34
#define PROW   (PDIM * CCH)
#define PIMG   (PDIM * PDIM * CCH)
#define ACT_BYTES (NIMG * PIMG)

#define BP_INTS (KSTEPS * 32 * 8)

__device__ __forceinline__ signed char sgn8f(float v) {
    return (signed char)((v > 0.0f) - (v < 0.0f));
}

__global__ void k_zero(int4* __restrict__ p) {
    int i = blockIdx.x * blockDim.x + threadIdx.x;
    p[i] = make_int4(0, 0, 0, 0);
}

__global__ void k_binarize_x(const float* __restrict__ x,
                             signed char* __restrict__ act) {
    int idx = blockIdx.x * blockDim.x + threadIdx.x;
    float v  = x[idx];
    int n  = idx >> 18;
    int c  = (idx >> 10) & 255;
    int h  = (idx >> 5) & 31;
    int w  = idx & 31;
    act[((n * (PDIM * PDIM) + (h + 1) * PDIM + (w + 1)) << 8) + c] = sgn8f(v);
}

__global__ void k_wmean(const float* __restrict__ w1,
                        const float* __restrict__ w2,
                        float* __restrict__ wmean) {
    __shared__ float red[256];
    int b = blockIdx.x;
    const float* w = (b < 256) ? w1 : w2;
    int co = b & 255;
    const float* p = w + co * KTOT;
    float s = 0.0f;
    for (int i = threadIdx.x; i < KTOT; i += 256) s += p[i];
    red[threadIdx.x] = s;
    __syncthreads();
    for (int off = 128; off > 0; off >>= 1) {
        if (threadIdx.x < off) red[threadIdx.x] += red[threadIdx.x + off];
        __syncthreads();
    }
    if (threadIdx.x == 0) wmean[b] = red[0] / (float)KTOT;
}

__global__ void k_pack_w(const float* __restrict__ w1,
                         const float* __restrict__ w2,
                         const float* __restrict__ wmean,
                         int* __restrict__ bp1, int* __restrict__ bp2) {
    int e = blockIdx.x * blockDim.x + threadIdx.x;
    const int PER = KSTEPS * NTILES * 32 * 8;
    int conv = e >= PER;
    int r = e - conv * PER;
    int v     = r & 7;
    int lane  = (r >> 3) & 31;
    int kstep = (r >> 8) % KSTEPS;
    int ntile = (r >> 8) / KSTEPS;
    const float* w = conv ? w2 : w1;
    int* bp        = conv ? bp2 : bp1;
    int co = ntile * 16 + (lane & 15);
    float mu = wmean[conv * 256 + co];
    int kb = ((lane >> 4) ? 16 : 0) + ((v >> 2) ? 32 : 0) + (v & 3) * 4;
    int word = 0;
    for (int b = 0; b < 4; ++b) {
        int kg  = kstep * 64 + kb + b;
        int tap = kg >> 8;
        int ci  = kg & 255;
        float wv = w[co * KTOT + ci * 9 + tap] - mu;
        int s = (wv > 0.0f) - (wv < 0.0f);
        word |= (s & 0xFF) << (8 * b);
    }
    bp[r] = word;
}

__global__ void k_scales(const float* a1, const float* g1, const float* be1,
                         const float* m1, const float* v1,
                         const float* a2, const float* g2, const float* be2,
                         const float* m2, const float* v2,
                         float* s1, float* b1, float* s2, float* b2) {
    int c = threadIdx.x;
    float i1 = g1[c] * rsqrtf(v1[c] + 1e-5f);
    s1[c] = a1[c] * i1;
    b1[c] = be1[c] - m1[c] * i1;
    float i2 = g2[c] * rsqrtf(v2[c] + 1e-5f);
    s2[c] = a2[c] * i2;
    b2[c] = be2[c] - m2[c] * i2;
}

template <bool STAGE2>
__global__ void __launch_bounds__(256, 2)
k_bingemm(const signed char* __restrict__ act,
          const int* __restrict__ bpack,
          const float* __restrict__ scale,
          const float* __restrict__ shift,
          const float* __restrict__ xin,
          float* __restrict__ out,
          signed char* __restrict__ actout) {
    __shared__ __align__(16) int ldsB[BP_INTS];

    const int lane  = threadIdx.x & 31;
    const int wv    = threadIdx.x >> 5;
    const int ntile = blockIdx.x & 15;
    const int mtile = (blockIdx.x >> 4) * 8 + wv;
    const int hi    = lane >> 4;

    const int* gB = bpack + ntile * BP_INTS;
#if __has_builtin(__builtin_amdgcn_global_load_async_to_lds_b128)
    for (int i = threadIdx.x; i < BP_INTS / 4; i += 256) {
        __builtin_amdgcn_global_load_async_to_lds_b128(
            (__attribute__((address_space(1))) gv4i*)(gB + i * 4),
            (__attribute__((address_space(3))) gv4i*)(ldsB + i * 4),
             0,  0);
    }
  #if __has_builtin(__builtin_amdgcn_s_wait_asynccnt)
    __builtin_amdgcn_s_wait_asynccnt(0);
  #else
    asm volatile("s_wait_asynccnt 0x0" ::: "memory");
  #endif
#else
    for (int i = threadIdx.x; i < BP_INTS / 4; i += 256) {
        ((int4*)ldsB)[i] = ((const int4*)gB)[i];
    }
#endif
    __syncthreads();

    const int p    = mtile * 16 + (lane & 15);
    const int nimg = p >> 10;
    const int hrow = (p >> 5) & 31;
    const int wcol = p & 31;

    const signed char* abasev[9];
    #pragma unroll
    for (int tap = 0; tap < 9; ++tap) {
        const int row = hrow + tap / 3;
        const int col = wcol + tap % 3;
        abasev[tap] = act + ((nimg * (PDIM * PDIM) + row * PDIM + col) << 8);
        __builtin_prefetch((const void*)abasev[tap], 0, 1);
    }

    v8i acc = {0, 0, 0, 0, 0, 0, 0, 0};

    #pragma unroll
    for (int tap = 0; tap < 9; ++tap) {
        const signed char* abase = abasev[tap];
        #pragma unroll
        for (int q = 0; q < 4; ++q) {
            const int kstep = tap * 4 + q;
            v8i b = *(const v8i*)&ldsB[kstep * 256 + lane * 8];
            const int cib = q * 64 + hi * 16;
            v8i a;
            #pragma unroll
            for (int v = 0; v < 8; ++v) {
                const int kl = ((v >> 1) & 1) * 8 + ((v >> 2) & 1) * 32 + (v & 1) * 4;
                a[v] = *(const int*)(abase + cib + kl);
            }
            acc = __builtin_amdgcn_wmma_i32_16x16x64_iu8(
                 true, a,  true, b, acc,
                 false,  false);
        }
    }

    const int ch = ntile * 16 + (lane & 15);
    const float sc = scale[ch];
    const float sh = shift[ch];
    #pragma unroll
    for (int r = 0; r < 8; ++r) {
        const int pix = mtile * 16 + r + hi * 8;
        const int n2 = pix >> 10;
        const int h2 = (pix >> 5) & 31;
        const int w2 = pix & 31;
        float y = (float)acc[r] * sc + sh;
        if (STAGE2) {
            const int idx = (((n2 << 8) + ch) << 10) + (h2 << 5) + w2;
            y += xin[idx];
            y = fminf(fmaxf(y, -1.0f), 1.0f);
            out[idx] = y;
        } else {
            actout[((n2 * (PDIM * PDIM) + (h2 + 1) * PDIM + (w2 + 1)) << 8) + ch]
                = sgn8f(y);
        }
    }
}

extern "C" void kernel_launch(void* const* d_in, const int* in_sizes, int n_in,
                              void* d_out, int out_size, void* d_ws, size_t ws_size,
                              hipStream_t stream) {
    const float* x   = (const float*)d_in[0];
    const float* w1  = (const float*)d_in[1];
    const float* a1  = (const float*)d_in[2];
    const float* g1  = (const float*)d_in[3];
    const float* be1 = (const float*)d_in[4];
    const float* mn1 = (const float*)d_in[5];
    const float* vr1 = (const float*)d_in[6];
    const float* w2  = (const float*)d_in[7];
    const float* a2  = (const float*)d_in[8];
    const float* g2  = (const float*)d_in[9];
    const float* be2 = (const float*)d_in[10];
    const float* mn2 = (const float*)d_in[11];
    const float* vr2 = (const float*)d_in[12];
    float* out = (float*)d_out;

    char* ws = (char*)d_ws;
    signed char* act1 = (signed char*)(ws);
    signed char* act2 = (signed char*)(ws + ACT_BYTES);
    int*   bp1   = (int*)(ws + 2 * (size_t)ACT_BYTES);
    int*   bp2   = (int*)(ws + 2 * (size_t)ACT_BYTES + 589824);
    float* wmean = (float*)(ws + 2 * (size_t)ACT_BYTES + 1179648);
    float* s1    = (float*)(ws + 2 * (size_t)ACT_BYTES + 1181696);
    float* b1    = s1 + 256;
    float* s2    = b1 + 256;
    float* b2    = s2 + 256;

    k_zero<<<(2 * ACT_BYTES / 16) / 256, 256, 0, stream>>>((int4*)ws);
    k_binarize_x<<<NPIX * CCH / 256, 256, 0, stream>>>(x, act1);
    k_wmean<<<512, 256, 0, stream>>>(w1, w2, wmean);
    k_pack_w<<<(2 * KSTEPS * NTILES * 32 * 8) / 256, 256, 0, stream>>>(
        w1, w2, wmean, bp1, bp2);
    k_scales<<<1, 256, 0, stream>>>(a1, g1, be1, mn1, vr1,
                                    a2, g2, be2, mn2, vr2, s1, b1, s2, b2);

    const int gblocks = (MTILES / 8) * NTILES;
    k_bingemm<false><<<gblocks, 256, 0, stream>>>(act1, bp1, s1, b1,
                                                  nullptr, nullptr, act2);
    k_bingemm<true><<<gblocks, 256, 0, stream>>>(act2, bp2, s2, b2,
                                                 x, out, nullptr);
}
